// WindowAttention_54279796687017
// MI455X (gfx1250) — matrix operand order repaired and checked by reading, not compiled, not run on hardware
//
#include <hip/hip_runtime.h>
#include <hip/hip_bf16.h>

// ---------------------------------------------------------------------------
// Swin window attention, fused, bf16 WMMA (gfx1250 / CDNA5, wave32)
//   B=8192 windows, N=49 (pad 64), DIM=192, HEADS=6, HEAD_DIM=32, NW=64
// ---------------------------------------------------------------------------

typedef unsigned short u16;
typedef __attribute__((ext_vector_type(16))) __bf16 v16bf;
typedef __attribute__((ext_vector_type(8)))  float  v8f;

#define NTOK   49
#define NP     64
#define DIMC   192
#define NHEAD  6
#define HD     32
#define NWIN   64
#define SCALE_C 0.17677669529663687f   // 32^-0.5

// float -> bf16 (round to nearest even), on raw bits
__device__ __forceinline__ u16 f2bf(float f) {
    unsigned u = __float_as_uint(f);
    u += 0x7FFFu + ((u >> 16) & 1u);
    return (u16)(u >> 16);
}

// A-operand tile 16(M) x 32(K) bf16, row-major storage with leading dim ld.
// ISA layout: lanes 0-15 row=lane, K in {lg*8+0..7, 16+lg*8+0..7} per lane half.
__device__ __forceinline__ v16bf load_a16x32(const u16* base, int ld, int lane) {
    const int lg = lane >> 4;
    const __bf16* p = (const __bf16*)(base + (lane & 15) * ld + lg * 8);
    v16bf a;
#pragma unroll
    for (int i = 0; i < 8; ++i) { a[i] = p[i]; a[i + 8] = p[16 + i]; }
    return a;
}

// B-operand tile 32(K) x 16(N) where B(k,n) = base[n*ld + k] (i.e. we hold B^T
// row-major).  ISA layout: lanes 0-15 col=lane, K in {lg*8+0..7, 16+lg*8+0..7} per lane half.
__device__ __forceinline__ v16bf load_bT32x16(const u16* base, int ld, int lane) {
    const int lg = lane >> 4;
    const __bf16* p = (const __bf16*)(base + (lane & 15) * ld + lg * 8);
    v16bf b;
#pragma unroll
    for (int i = 0; i < 8; ++i) { b[i] = p[i]; b[i + 8] = p[16 + i]; }
    return b;
}

__device__ __forceinline__ v8f wmma_bf16(v16bf a, v16bf b, v8f c) {
    return __builtin_amdgcn_wmma_f32_16x16x32_bf16(false, a, false, b,
                                                   (short)0, c, false, false);
}

// --------------------------- prep kernels ----------------------------------

__global__ void prep_weights(const float* __restrict__ qkv_w,
                             const float* __restrict__ proj_w,
                             u16* __restrict__ qkvw_bf,
                             u16* __restrict__ projw_bf) {
    int i = blockIdx.x * blockDim.x + threadIdx.x;
    const int n1 = 3 * DIMC * DIMC;   // 576*192
    const int n2 = DIMC * DIMC;       // 192*192
    if (i < n1)            qkvw_bf[i]      = f2bf(qkv_w[i]);
    else if (i < n1 + n2)  projw_bf[i - n1] = f2bf(proj_w[i - n1]);
}

// biasP[h][q(64)][k(64)]: rel-position bias gathered per head, padding columns
// forced to -1e30 so softmax zeroes them; padding rows kept finite.
__global__ void prep_bias(const float* __restrict__ rel_table,
                          const int* __restrict__ rel_index,
                          float* __restrict__ biasP) {
    int i = blockIdx.x * blockDim.x + threadIdx.x;
    if (i >= NHEAD * NP * NP) return;
    int h = i / (NP * NP);
    int q = (i / NP) % NP;
    int k = i % NP;
    float v;
    if (k >= NTOK)      v = -1.0e30f;
    else if (q >= NTOK) v = 0.0f;
    else                v = rel_table[rel_index[q * NTOK + k] * NHEAD + h];
    biasP[i] = v;
}

// --------------------------- fused attention -------------------------------
// LDS map (dynamic, 147456 B):
//   [0,      24576)  xb  [64][192] bf16   (aliased later by outb)
//   [24576,  49152)  Qs  [6][64][32] bf16 (pre-scaled)
//   [49152,  73728)  Ks  [6][64][32] bf16
//   [73728,  98304)  Vt  [6][32][64] bf16 (transposed V)
//   [98304, 147456)  Pb  [6][64][64] bf16 (softmax probs)

__global__ __launch_bounds__(256)
void swin_attn_kernel(const float* __restrict__ x,
                      const float* __restrict__ mask,
                      const float* __restrict__ qkv_b,
                      const float* __restrict__ proj_b,
                      const u16*  __restrict__ qkvw,    // [576][192] bf16
                      const u16*  __restrict__ projw,   // [192][192] bf16
                      const float* __restrict__ biasP,  // [6][64][64]
                      float* __restrict__ out) {
    extern __shared__ char smem[];
    u16* xb   = (u16*)smem;                    // [64][192]
    u16* outb = xb;                            // alias (xb dead after QKV)
    u16* Qs   = (u16*)(smem + 24576);          // [6][64][32]
    u16* Ks   = (u16*)(smem + 49152);          // [6][64][32]
    u16* Vt   = (u16*)(smem + 73728);          // [6][32][64]
    u16* Pb   = (u16*)(smem + 98304);          // [6][64][64]

    const int b    = blockIdx.x;
    const int tid  = threadIdx.x;
    const int lane = tid & 31;
    const int wave = tid >> 5;
    const int w    = b & (NWIN - 1);           // window-mask index = b % 64
    const int col  = lane & 15;
    const int lg   = lane >> 4;

    const v8f vzero = {0.f, 0.f, 0.f, 0.f, 0.f, 0.f, 0.f, 0.f};

    // ---- phase 1: stage x -> bf16 LDS, zero-pad rows 49..63 ----
    const float* xw = x + (size_t)b * NTOK * DIMC;
    for (int c = tid; c < NP * DIMC / 4; c += 256) {
        int e   = c * 4;
        int row = e / DIMC, cc = e % DIMC;
        float4 v = make_float4(0.f, 0.f, 0.f, 0.f);
        if (row < NTOK) v = *(const float4*)(xw + row * DIMC + cc);
        u16* d = xb + row * DIMC + cc;
        d[0] = f2bf(v.x); d[1] = f2bf(v.y); d[2] = f2bf(v.z); d[3] = f2bf(v.w);
    }
    __syncthreads();

    // ---- phase 2: QKV projection, 36 column tiles of [64 x 16] ----
    for (int t = wave; t < 36; t += 8) {
        const int n0   = t * 16;
        const float bc = qkv_b[n0 + col];
        v8f acc[4] = {vzero, vzero, vzero, vzero};
        for (int ks = 0; ks < 6; ++ks) {
            v16bf bt = load_bT32x16(qkvw + (size_t)n0 * DIMC + ks * 32, DIMC, lane);
#pragma unroll
            for (int m = 0; m < 4; ++m) {
                v16bf at = load_a16x32(xb + m * 16 * DIMC + ks * 32, DIMC, lane);
                acc[m] = wmma_bf16(at, bt, acc[m]);
            }
        }
        const int which = n0 / DIMC;           // 0=q 1=k 2=v
        const int h     = (n0 % DIMC) / HD;
        const int d0    = n0 % HD;             // 0 or 16
#pragma unroll
        for (int m = 0; m < 4; ++m) {
#pragma unroll
            for (int j = 0; j < 8; ++j) {
                int row = m * 16 + lg * 8 + j;
                float v = acc[m][j] + bc;
                if (which == 0)      Qs[(h * NP + row) * HD + d0 + col] = f2bf(v * SCALE_C);
                else if (which == 1) Ks[(h * NP + row) * HD + d0 + col] = f2bf(v);
                else                 Vt[(h * HD + d0 + col) * NP + row] = f2bf(v);
            }
        }
    }
    __syncthreads();

    // ---- phase 3: Q K^T + bias + mask + softmax (in registers) ----
    for (int u = wave; u < 24; u += 8) {
        const int h  = u / 4;
        const int tm = u % 4;
        const u16* Qh = Qs + h * NP * HD;
        const u16* Kh = Ks + h * NP * HD;
        v16bf aq = load_a16x32(Qh + tm * 16 * HD, HD, lane);
        v8f at[4];
#pragma unroll
        for (int tn = 0; tn < 4; ++tn) {
            v16bf bk = load_bT32x16(Kh + tn * 16 * HD, HD, lane);
            at[tn] = wmma_bf16(aq, bk, vzero);
        }
        const float* bph = biasP + h * NP * NP;
#pragma unroll
        for (int j = 0; j < 8; ++j) {
            const int q = tm * 16 + lg * 8 + j;
            float vals[4];
            float vmax = -3.0e38f;
#pragma unroll
            for (int tn = 0; tn < 4; ++tn) {
                const int k = tn * 16 + col;
                float bias = bph[q * NP + k];
                float mv = (q < NTOK && k < NTOK)
                             ? mask[(w * NTOK + q) * NTOK + k] : 0.f;
                float v = at[tn][j] + bias + mv;
                vals[tn] = v;
                vmax = fmaxf(vmax, v);
            }
#pragma unroll
            for (int s = 1; s < 16; s <<= 1)
                vmax = fmaxf(vmax, __shfl_xor(vmax, s, 32));
            float sum = 0.f;
#pragma unroll
            for (int tn = 0; tn < 4; ++tn) {
                vals[tn] = __expf(vals[tn] - vmax);
                sum += vals[tn];
            }
#pragma unroll
            for (int s = 1; s < 16; s <<= 1)
                sum += __shfl_xor(sum, s, 32);
            const float inv = 1.f / sum;
            u16* prow = Pb + (h * NP + q) * NP;
#pragma unroll
            for (int tn = 0; tn < 4; ++tn)
                prow[tn * 16 + col] = f2bf(vals[tn] * inv);
        }
    }
    __syncthreads();

    // ---- phase 4: P @ V -> outb [64][192] bf16 ----
    for (int u = wave; u < 24; u += 8) {
        const int h  = u / 4;
        const int tm = u % 4;
        const u16* Ph = Pb + h * NP * NP;
        const u16* Vh = Vt + h * HD * NP;
        v8f acc[2] = {vzero, vzero};
#pragma unroll
        for (int ks = 0; ks < 2; ++ks) {
            v16bf ap = load_a16x32(Ph + tm * 16 * NP + ks * 32, NP, lane);
#pragma unroll
            for (int tn = 0; tn < 2; ++tn) {
                v16bf bv = load_bT32x16(Vh + tn * 16 * NP + ks * 32, NP, lane);
                acc[tn] = wmma_bf16(ap, bv, acc[tn]);
            }
        }
#pragma unroll
        for (int tn = 0; tn < 2; ++tn)
#pragma unroll
            for (int j = 0; j < 8; ++j) {
                int row = tm * 16 + lg * 8 + j;
                outb[row * DIMC + h * HD + tn * 16 + col] = f2bf(acc[tn][j]);
            }
    }
    __syncthreads();

    // ---- phase 5: output projection, store fp32 rows q<49 ----
    float* yw = out + (size_t)b * NTOK * DIMC;
    for (int t = wave; t < 48; t += 8) {
        const int tn = t % 12, tm = t / 12;
        const int n0 = tn * 16;
        v8f acc = vzero;
        for (int ks = 0; ks < 6; ++ks) {
            v16bf a  = load_a16x32(outb + tm * 16 * DIMC + ks * 32, DIMC, lane);
            v16bf bt = load_bT32x16(projw + (size_t)n0 * DIMC + ks * 32, DIMC, lane);
            acc = wmma_bf16(a, bt, acc);
        }
        const float pb = proj_b[n0 + col];
#pragma unroll
        for (int j = 0; j < 8; ++j) {
            int row = tm * 16 + lg * 8 + j;
            if (row < NTOK) yw[row * DIMC + n0 + col] = acc[j] + pb;
        }
    }
}

// --------------------------- launcher --------------------------------------

extern "C" void kernel_launch(void* const* d_in, const int* in_sizes, int n_in,
                              void* d_out, int out_size, void* d_ws, size_t ws_size,
                              hipStream_t stream) {
    const float* x         = (const float*)d_in[0];
    const float* mask      = (const float*)d_in[1];
    const float* qkv_w     = (const float*)d_in[2];
    const float* qkv_b     = (const float*)d_in[3];
    const float* proj_w    = (const float*)d_in[4];
    const float* proj_b    = (const float*)d_in[5];
    const float* rel_table = (const float*)d_in[6];
    const int*   rel_index = (const int*)d_in[7];
    float* out = (float*)d_out;

    // workspace: qkvw bf16 (221184 B) | projw bf16 (73728 B) | biasP (98304 B)
    char* ws = (char*)d_ws;
    u16*   qkvw_bf  = (u16*)ws;
    u16*   projw_bf = (u16*)(ws + 221184);
    float* biasP    = (float*)(ws + 221184 + 73728);

    const int nconv = 3 * DIMC * DIMC + DIMC * DIMC;
    prep_weights<<<(nconv + 255) / 256, 256, 0, stream>>>(qkv_w, proj_w,
                                                          qkvw_bf, projw_bf);
    prep_bias<<<(NHEAD * NP * NP + 255) / 256, 256, 0, stream>>>(rel_table,
                                                                 rel_index, biasP);

    const size_t lds_bytes = 147456;  // 144 KB -> 2 workgroups / WGP (320 KB)
    (void)hipFuncSetAttribute((const void*)swin_attn_kernel,
                              hipFuncAttributeMaxDynamicSharedMemorySize,
                              (int)lds_bytes);
    swin_attn_kernel<<<8192, 256, lds_bytes, stream>>>(x, mask, qkv_b, proj_b,
                                                       qkvw_bf, projw_bf,
                                                       biasP, out);
}
